// SubAdjacentAttention_82549271429741
// MI455X (gfx1250) — matrix operand order repaired and checked by reading, not compiled, not run on hardware
//
#include <hip/hip_runtime.h>

typedef __bf16 bf16_t;
typedef __attribute__((ext_vector_type(16))) __bf16 v16bf;
typedef __attribute__((ext_vector_type(8)))  __bf16 v8bf;
typedef __attribute__((ext_vector_type(8)))  float  v8f;

#define D_MODEL 512
#define N_HEADS 8
#define HEAD_D  64
#define SEQ     2048
#define BATCH   4
#define TOK     (BATCH * SEQ)   // 8192
#define WIN     5

// ---------------------------------------------------------------------------
// helpers
// ---------------------------------------------------------------------------
__device__ __forceinline__ v16bf load_frag2(const bf16_t* p0, const bf16_t* p1) {
    union { v16bf v; v8bf h[2]; } u;
    u.h[0] = *reinterpret_cast<const v8bf*>(p0);
    u.h[1] = *reinterpret_cast<const v8bf*>(p1);
    return u.v;
}

__global__ __launch_bounds__(256) void f32_to_bf16_kernel(const float* __restrict__ in,
                                                          bf16_t* __restrict__ out, int n) {
    int i = blockIdx.x * 256 + threadIdx.x;
    if (i < n) out[i] = (bf16_t)in[i];
}

// ---------------------------------------------------------------------------
// Generic bf16 GEMM:  Out[M,N] = act(A[M,K] @ W[K,N] + bias[N])
// block tile 128x128, BK=32, 8 waves (2 along M x 4 along N), wave tile 64x32
// ---------------------------------------------------------------------------
#define BM 128
#define BN 128
#define BK 32
#define LDA_S 40   // LDS row stride (elements) for As [m][k]  (80B, 16B aligned)
#define LDB_S 40   // LDS row stride for Bs stored transposed [n][k]

template <bool RELU, bool OUT_BF16>
__global__ __launch_bounds__(256) void gemm_bf16_kernel(const bf16_t* __restrict__ A,
                                                        const bf16_t* __restrict__ W,
                                                        const float* __restrict__ bias,
                                                        void* __restrict__ Out,
                                                        int M, int N, int K) {
    __shared__ bf16_t As[BM * LDA_S];
    __shared__ bf16_t Bs[BN * LDB_S];

    const int tid  = threadIdx.x;
    const int lane = tid & 31;
    const int w    = tid >> 5;
    const int wm   = w >> 2;        // 0..1
    const int wn   = w & 3;         // 0..3
    const int m0   = blockIdx.y * BM;
    const int n0   = blockIdx.x * BN;

    v8f acc[4][2];
#pragma unroll
    for (int mt = 0; mt < 4; mt++)
#pragma unroll
        for (int nt = 0; nt < 2; nt++)
            acc[mt][nt] = (v8f){0.f, 0.f, 0.f, 0.f, 0.f, 0.f, 0.f, 0.f};

    const int a_kp = tid & 15;      // u32 pair index within a 32-wide k row
    const int a_m0 = tid >> 4;      // base row, step 16
    const int b_np = tid & 63;      // u32 pair index along N
    const int b_k0 = tid >> 6;      // base k row, step 4

    const int row16 = lane & 15;
    const int akb   = (lane < 16) ? 0 : 8;    // A-matrix lane K base
    const int bkb   = (lane < 16) ? 0 : 8;    // B-matrix lane K base

    for (int ks = 0; ks < K; ks += BK) {
        __syncthreads();
        // ---- stage A tile [128 x 32] (bf16, row major) ----
#pragma unroll
        for (int j = 0; j < 8; j++) {
            int m = a_m0 + 16 * j;
            unsigned d = *(reinterpret_cast<const unsigned*>(A + (size_t)(m0 + m) * K + ks) + a_kp);
            *reinterpret_cast<unsigned*>(&As[m * LDA_S + 2 * a_kp]) = d;
        }
        // ---- stage B tile transposed: Bs[n][k] = W[ks+k][n0+n] ----
#pragma unroll
        for (int j = 0; j < 8; j++) {
            int k = b_k0 + 4 * j;
            unsigned d = *(reinterpret_cast<const unsigned*>(W + (size_t)(ks + k) * N + n0) + b_np);
            union { unsigned u; bf16_t b[2]; } t; t.u = d;
            int n = 2 * b_np;
            Bs[n * LDB_S + k]       = t.b[0];
            Bs[(n + 1) * LDB_S + k] = t.b[1];
        }
        __syncthreads();

        // ---- fragments + 8 WMMAs ----
        v16bf afr[4], bfr[2];
#pragma unroll
        for (int mt = 0; mt < 4; mt++) {
            const bf16_t* base = &As[(wm * 64 + mt * 16 + row16) * LDA_S];
            afr[mt] = load_frag2(base + akb, base + 16 + akb);
        }
#pragma unroll
        for (int nt = 0; nt < 2; nt++) {
            const bf16_t* base = &Bs[(wn * 32 + nt * 16 + row16) * LDB_S];
            bfr[nt] = load_frag2(base + bkb, base + 16 + bkb);
        }
#pragma unroll
        for (int mt = 0; mt < 4; mt++)
#pragma unroll
            for (int nt = 0; nt < 2; nt++)
                acc[mt][nt] = __builtin_amdgcn_wmma_f32_16x16x32_bf16(
                    false, afr[mt], false, bfr[nt], (short)0, acc[mt][nt], false, false);
    }

    // ---- epilogue: bias, activation, store ----
    const int rofs = (lane < 16) ? 0 : 8;
#pragma unroll
    for (int nt = 0; nt < 2; nt++) {
        int col = n0 + wn * 32 + nt * 16 + row16;
        float bv = bias[col];
#pragma unroll
        for (int mt = 0; mt < 4; mt++) {
#pragma unroll
            for (int i = 0; i < 8; i++) {
                int r = m0 + wm * 64 + mt * 16 + i + rofs;
                float v = acc[mt][nt][i] + bv;
                if (RELU) v = v > 0.f ? v : 0.f;
                if (OUT_BF16)
                    ((bf16_t*)Out)[(size_t)r * N + col] = (bf16_t)v;
                else
                    ((float*)Out)[(size_t)r * N + col] = v;
            }
        }
    }
}

// ---------------------------------------------------------------------------
// Flash attention with INVERSE sliding-window mask (|i-j| <= WIN excluded).
// grid: (SEQ/128, BATCH*N_HEADS), 256 threads (8 waves x 16 queries each)
// ---------------------------------------------------------------------------
#define KBLK 64
#define VLD  72   // LDS row stride (elements); 144B, 16B aligned

__global__ __launch_bounds__(256) void flash_attn_kernel(const bf16_t* __restrict__ Q,
                                                         const bf16_t* __restrict__ Km,
                                                         const bf16_t* __restrict__ V,
                                                         bf16_t* __restrict__ O) {
    __shared__ bf16_t Vt[HEAD_D * VLD];        // V^T tile: [channel][key]
    __shared__ bf16_t Pb[8 * 16 * VLD];        // per-wave probabilities [16][VLD]

    const int tid  = threadIdx.x;
    const int lane = tid & 31;
    const int w    = tid >> 5;
    const int bh   = blockIdx.y;
    const size_t headoff = (size_t)(bh >> 3) * SEQ * D_MODEL + (size_t)(bh & 7) * HEAD_D;
    const int q0 = blockIdx.x * 128 + w * 16;

    const int row16 = lane & 15;
    const int rofs  = (lane < 16) ? 0 : 8;
    const int akb   = (lane < 16) ? 0 : 8;
    const int bkb   = (lane < 16) ? 0 : 8;

    // Q fragments (A-matrix, 16 rows x 64 K): 2 k-steps, loaded once from global
    v16bf qf[2];
    {
        const bf16_t* qrow = Q + headoff + (size_t)(q0 + row16) * D_MODEL;
        qf[0] = load_frag2(qrow + akb,      qrow + 16 + akb);
        qf[1] = load_frag2(qrow + 32 + akb, qrow + 48 + akb);
    }

    v8f o[4];
#pragma unroll
    for (int ct = 0; ct < 4; ct++) o[ct] = (v8f){0.f, 0.f, 0.f, 0.f, 0.f, 0.f, 0.f, 0.f};
    float mrow[8], lrow[8];
#pragma unroll
    for (int i = 0; i < 8; i++) { mrow[i] = -3.0e38f; lrow[i] = 0.f; }

    bf16_t* Pw = &Pb[w * 16 * VLD];

    for (int kb = 0; kb < SEQ; kb += KBLK) {
        // ---- cooperatively stage V^T[ch][key] for 64 keys ----
        __syncthreads();
        {
            int key = tid >> 2;           // 0..63
            int cb  = (tid & 3) * 16;     // channel base
            const uint4* vsrc = reinterpret_cast<const uint4*>(
                V + headoff + (size_t)(kb + key) * D_MODEL + cb);
            uint4 d0 = vsrc[0], d1 = vsrc[1];
            unsigned uu[8] = {d0.x, d0.y, d0.z, d0.w, d1.x, d1.y, d1.z, d1.w};
#pragma unroll
            for (int j = 0; j < 8; j++) {
                union { unsigned u; bf16_t b[2]; } t; t.u = uu[j];
                int ch = cb + 2 * j;
                Vt[ch * VLD + key]       = t.b[0];
                Vt[(ch + 1) * VLD + key] = t.b[1];
            }
        }
        __syncthreads();

        // ---- scores: 4 key sub-tiles of 16; K^T B-frags straight from global ----
        v8f s[4];
#pragma unroll
        for (int kt = 0; kt < 4; kt++) {
            const bf16_t* krow = Km + headoff + (size_t)(kb + kt * 16 + row16) * D_MODEL;
            v16bf kf0 = load_frag2(krow + bkb,      krow + 16 + bkb);
            v16bf kf1 = load_frag2(krow + 32 + bkb, krow + 48 + bkb);
            v8f c = (v8f){0.f, 0.f, 0.f, 0.f, 0.f, 0.f, 0.f, 0.f};
            c = __builtin_amdgcn_wmma_f32_16x16x32_bf16(false, qf[0], false, kf0, (short)0, c, false, false);
            c = __builtin_amdgcn_wmma_f32_16x16x32_bf16(false, qf[1], false, kf1, (short)0, c, false, false);
            s[kt] = c;
        }

        // ---- scale + inverse-window mask ----
#pragma unroll
        for (int kt = 0; kt < 4; kt++) {
            int kcol = kb + kt * 16 + row16;
#pragma unroll
            for (int i = 0; i < 8; i++) {
                int qr = q0 + i + rofs;
                float v = s[kt][i] * 0.125f;          // 1/sqrt(64)
                int dd = qr - kcol; if (dd < 0) dd = -dd;
                if (dd <= WIN) v = -1e9f;
                s[kt][i] = v;
            }
        }

        // ---- online softmax: block row-max via 16-lane butterflies ----
        float alpha[8];
#pragma unroll
        for (int i = 0; i < 8; i++) {
            float v = fmaxf(fmaxf(s[0][i], s[1][i]), fmaxf(s[2][i], s[3][i]));
#pragma unroll
            for (int msk = 1; msk < 16; msk <<= 1) v = fmaxf(v, __shfl_xor(v, msk, 32));
            float mnew = fmaxf(mrow[i], v);
            alpha[i] = __expf(mrow[i] - mnew);
            mrow[i] = mnew;
            lrow[i] *= alpha[i];
        }
#pragma unroll
        for (int ct = 0; ct < 4; ct++)
#pragma unroll
            for (int i = 0; i < 8; i++) o[ct][i] *= alpha[i];

        // ---- p = exp(s - m): write P (C-layout -> LDS), accumulate row sums ----
#pragma unroll
        for (int kt = 0; kt < 4; kt++) {
#pragma unroll
            for (int i = 0; i < 8; i++) {
                float p = __expf(s[kt][i] - mrow[i]);
                s[kt][i] = p;
                Pw[(i + rofs) * VLD + kt * 16 + row16] = (bf16_t)p;
            }
        }
#pragma unroll
        for (int i = 0; i < 8; i++) {
            float rs = s[0][i] + s[1][i] + s[2][i] + s[3][i];
#pragma unroll
            for (int msk = 1; msk < 16; msk <<= 1) rs += __shfl_xor(rs, msk, 32);
            lrow[i] += rs;
        }

        // ---- O += P @ V  (P A-frags from per-wave LDS, V^T B-frags from LDS) ----
        v16bf pf[2];
        {
            const bf16_t* prow = &Pw[row16 * VLD];
            pf[0] = load_frag2(prow + akb,      prow + 16 + akb);
            pf[1] = load_frag2(prow + 32 + akb, prow + 48 + akb);
        }
#pragma unroll
        for (int ct = 0; ct < 4; ct++) {
            const bf16_t* vrow = &Vt[(ct * 16 + row16) * VLD];
            v16bf vf0 = load_frag2(vrow + bkb,      vrow + 16 + bkb);
            v16bf vf1 = load_frag2(vrow + 32 + bkb, vrow + 48 + bkb);
            o[ct] = __builtin_amdgcn_wmma_f32_16x16x32_bf16(false, pf[0], false, vf0, (short)0, o[ct], false, false);
            o[ct] = __builtin_amdgcn_wmma_f32_16x16x32_bf16(false, pf[1], false, vf1, (short)0, o[ct], false, false);
        }
    }

    // ---- normalize and store ----
#pragma unroll
    for (int ct = 0; ct < 4; ct++) {
#pragma unroll
        for (int i = 0; i < 8; i++) {
            int qr = q0 + i + rofs;
            float v = o[ct][i] / lrow[i];
            O[headoff + (size_t)qr * D_MODEL + ct * 16 + row16] = (bf16_t)v;
        }
    }
}

// ---------------------------------------------------------------------------
// launch
// ---------------------------------------------------------------------------
extern "C" void kernel_launch(void* const* d_in, const int* in_sizes, int n_in,
                              void* d_out, int out_size, void* d_ws, size_t ws_size,
                              hipStream_t stream) {
    (void)in_sizes; (void)n_in; (void)out_size; (void)ws_size;

    const float* x  = (const float*)d_in[0];
    const float* W1 = (const float*)d_in[1];
    const float* b1 = (const float*)d_in[2];
    const float* W2 = (const float*)d_in[3];
    const float* b2 = (const float*)d_in[4];
    const float* Wq = (const float*)d_in[5];
    const float* bq = (const float*)d_in[6];
    const float* Wk = (const float*)d_in[7];
    const float* bk = (const float*)d_in[8];
    const float* Wv = (const float*)d_in[9];
    const float* bv = (const float*)d_in[10];
    const float* Wo = (const float*)d_in[11];
    const float* bo = (const float*)d_in[12];

    char* ws = (char*)d_ws;
    size_t off = 0;
    auto alloc = [&](size_t bytes) { void* p = ws + off; off += (bytes + 255) & ~(size_t)255; return p; };

    bf16_t* x_bf   = (bf16_t*)alloc((size_t)TOK * D_MODEL * 2);
    bf16_t* W1_bf  = (bf16_t*)alloc((size_t)D_MODEL * 2 * D_MODEL * 2);
    bf16_t* W2_bf  = (bf16_t*)alloc((size_t)2 * D_MODEL * D_MODEL * 2);
    bf16_t* Wq_bf  = (bf16_t*)alloc((size_t)D_MODEL * D_MODEL * 2);
    bf16_t* Wk_bf  = (bf16_t*)alloc((size_t)D_MODEL * D_MODEL * 2);
    bf16_t* Wv_bf  = (bf16_t*)alloc((size_t)D_MODEL * D_MODEL * 2);
    bf16_t* Wo_bf  = (bf16_t*)alloc((size_t)D_MODEL * D_MODEL * 2);
    bf16_t* H_bf   = (bf16_t*)alloc((size_t)TOK * 2 * D_MODEL * 2);
    bf16_t* xm_bf  = (bf16_t*)alloc((size_t)TOK * D_MODEL * 2);
    bf16_t* Q_bf   = (bf16_t*)alloc((size_t)TOK * D_MODEL * 2);
    bf16_t* K_bf   = (bf16_t*)alloc((size_t)TOK * D_MODEL * 2);
    bf16_t* V_bf   = (bf16_t*)alloc((size_t)TOK * D_MODEL * 2);
    bf16_t* AO_bf  = (bf16_t*)alloc((size_t)TOK * D_MODEL * 2);

    auto cvt = [&](const float* src, bf16_t* dst, int n) {
        f32_to_bf16_kernel<<<(n + 255) / 256, 256, 0, stream>>>(src, dst, n);
    };
    cvt(x,  x_bf,  TOK * D_MODEL);
    cvt(W1, W1_bf, D_MODEL * 2 * D_MODEL);
    cvt(W2, W2_bf, 2 * D_MODEL * D_MODEL);
    cvt(Wq, Wq_bf, D_MODEL * D_MODEL);
    cvt(Wk, Wk_bf, D_MODEL * D_MODEL);
    cvt(Wv, Wv_bf, D_MODEL * D_MODEL);
    cvt(Wo, Wo_bf, D_MODEL * D_MODEL);

    const dim3 blk(256);
    // H = relu(x @ W1 + b1)   [8192 x 1024]
    gemm_bf16_kernel<true, true><<<dim3((2 * D_MODEL) / BN, TOK / BM), blk, 0, stream>>>(
        x_bf, W1_bf, b1, H_bf, TOK, 2 * D_MODEL, D_MODEL);
    // xm = H @ W2 + b2        [8192 x 512]
    gemm_bf16_kernel<false, true><<<dim3(D_MODEL / BN, TOK / BM), blk, 0, stream>>>(
        H_bf, W2_bf, b2, xm_bf, TOK, D_MODEL, 2 * D_MODEL);
    // Q/K/V projections
    gemm_bf16_kernel<false, true><<<dim3(D_MODEL / BN, TOK / BM), blk, 0, stream>>>(
        xm_bf, Wq_bf, bq, Q_bf, TOK, D_MODEL, D_MODEL);
    gemm_bf16_kernel<false, true><<<dim3(D_MODEL / BN, TOK / BM), blk, 0, stream>>>(
        xm_bf, Wk_bf, bk, K_bf, TOK, D_MODEL, D_MODEL);
    gemm_bf16_kernel<false, true><<<dim3(D_MODEL / BN, TOK / BM), blk, 0, stream>>>(
        xm_bf, Wv_bf, bv, V_bf, TOK, D_MODEL, D_MODEL);
    // flash attention with inverse sliding-window mask
    flash_attn_kernel<<<dim3(SEQ / 128, BATCH * N_HEADS), blk, 0, stream>>>(
        Q_bf, K_bf, V_bf, AO_bf);
    // final projection -> f32 output
    gemm_bf16_kernel<false, false><<<dim3(D_MODEL / BN, TOK / BM), blk, 0, stream>>>(
        AO_bf, Wo_bf, bo, (float*)d_out, TOK, D_MODEL, D_MODEL);
}
